// TextGuideAttention_46660524704010
// MI455X (gfx1250) — matrix operand order repaired and checked by reading, not compiled, not run on hardware
//
#include <hip/hip_runtime.h>
#include <hip/hip_bf16.h>

// ---------------------------------------------------------------------------
// TextGuideAttention block for MI455X (gfx1250), bf16 WMMA + f32 accumulate.
// B=4, S=4096, D=256, D_FF=1024. All GEMMs on v_wmma_f32_16x16x32_bf16.
// Data movement: b128 global loads, bf16 LDS staging, vector LDS reads.
// ---------------------------------------------------------------------------

#define B_    4
#define S_    4096
#define D_    256
#define DFF_  1024
#define WIN_  2
#define KTOP_ (S_/2 + 2*WIN_)     /* 2052 */
#define NTOK_ (B_*S_)             /* 16384 */
#define NTOT_ (NTOK_*D_)          /* 4194304 */

typedef __attribute__((ext_vector_type(16))) __bf16 v16bf;
typedef __attribute__((ext_vector_type(8)))  float  v8f;

static __device__ __forceinline__ v8f wmma_bf16(v16bf a, v16bf b, v8f c) {
  // D = A(16x32) * B(32x16) + C, f32 accumulate
  return __builtin_amdgcn_wmma_f32_16x16x32_bf16(false, a, false, b,
                                                 (short)0, c, false, false);
}

// A-matrix (16x32 bf16) element j (0..15) -> K index, as built by load_afrag:
//   j=0..7  -> K = half*16 + j
//   j=8..15 -> K = half*16 + j
// B-matrix (32x16 bf16): lane n = lane&15, element j -> K = half*16 + j.
// C/D (16x16 f32): lane holds (M = i + half*8, N = lane&15) in c[i].

// Build B-fragment from 16 contiguous floats (16B-aligned): 4x b128 loads.
static __device__ __forceinline__ v16bf load_bfrag16(const float* __restrict__ p) {
  const float4* q = (const float4*)p;
  float4 f0 = q[0], f1 = q[1], f2 = q[2], f3 = q[3];
  v16bf r;
  r[0]=(__bf16)f0.x; r[1]=(__bf16)f0.y; r[2]=(__bf16)f0.z; r[3]=(__bf16)f0.w;
  r[4]=(__bf16)f1.x; r[5]=(__bf16)f1.y; r[6]=(__bf16)f1.z; r[7]=(__bf16)f1.w;
  r[8]=(__bf16)f2.x; r[9]=(__bf16)f2.y; r[10]=(__bf16)f2.z; r[11]=(__bf16)f2.w;
  r[12]=(__bf16)f3.x; r[13]=(__bf16)f3.y; r[14]=(__bf16)f3.z; r[15]=(__bf16)f3.w;
  return r;
}

// Build A-fragment: p points at (row + kbase + half*16); runs p[0..15].
static __device__ __forceinline__ v16bf load_afrag(const float* __restrict__ p) {
  const float4* q = (const float4*)p;
  float4 f0 = q[0], f1 = q[1], f2 = q[2], f3 = q[3];
  v16bf r;
  r[0]=(__bf16)f0.x; r[1]=(__bf16)f0.y; r[2]=(__bf16)f0.z; r[3]=(__bf16)f0.w;
  r[4]=(__bf16)f1.x; r[5]=(__bf16)f1.y; r[6]=(__bf16)f1.z; r[7]=(__bf16)f1.w;
  r[8]=(__bf16)f2.x; r[9]=(__bf16)f2.y; r[10]=(__bf16)f2.z; r[11]=(__bf16)f2.w;
  r[12]=(__bf16)f3.x; r[13]=(__bf16)f3.y; r[14]=(__bf16)f3.z; r[15]=(__bf16)f3.w;
  return r;
}

static __device__ __forceinline__ unsigned pack2(float a, float b) {
  __bf16 x = (__bf16)a, y = (__bf16)b;
  unsigned short hx, hy;
  __builtin_memcpy(&hx, &x, 2); __builtin_memcpy(&hy, &y, 2);
  return (unsigned)hx | ((unsigned)hy << 16);
}

// ---------------------------------------------------------------------------
// GEMM: Out[M,N] = X[M,K] @ W[N,K]^T + bias[N]
// One 16x64 tile per wave: 4 accumulators reuse one A fragment.
// ---------------------------------------------------------------------------
__global__ void __launch_bounds__(128)
gemm_bias_wmma(const float* __restrict__ X, const float* __restrict__ W,
               const float* __restrict__ bias, float* __restrict__ Out,
               int M, int N, int K) {
  int wid  = (blockIdx.x * blockDim.x + threadIdx.x) >> 5;
  int lane = threadIdx.x & 31;
  int half = lane >> 4, ln = lane & 15;
  int tiles_n = N >> 6;                       // 64-wide N tiles
  int m0 = (wid / tiles_n) << 4;
  int n0 = (wid % tiles_n) << 6;
  if (m0 >= M) return;                        // wave-uniform, EXEC stays all-1
  v8f acc[4] = {{}, {}, {}, {}};
  const float* Xr = X + (size_t)(m0 + ln) * K;
  for (int c = 0; c < K; c += 32) {
    __builtin_prefetch(Xr + c + 128, 0, 3);
    v16bf a = load_afrag(Xr + c + half*16);
#pragma unroll
    for (int t = 0; t < 4; ++t) {
      v16bf b = load_bfrag16(W + (size_t)(n0 + t*16 + ln) * K + c + half*16);
      acc[t] = wmma_bf16(a, b, acc[t]);
    }
  }
#pragma unroll
  for (int t = 0; t < 4; ++t) {
    float bn = bias[n0 + t*16 + ln];
#pragma unroll
    for (int i = 0; i < 8; ++i)
      Out[(size_t)(m0 + i + half*8) * N + n0 + t*16 + ln] = acc[t][i] + bn;
  }
}

// ---------------------------------------------------------------------------
// Column prefix sums of k over the sequence dim: P[b,s,d] = sum_{j<=s} k[b,j,d]
// ---------------------------------------------------------------------------
__global__ void prefix_kernel(const float* __restrict__ k, float* __restrict__ P) {
  int id = blockIdx.x * blockDim.x + threadIdx.x;
  if (id >= B_*D_) return;
  int b = id / D_, d = id % D_;
  const float* kp = k + (size_t)b * S_ * D_ + d;
  float*       pp = P + (size_t)b * S_ * D_ + d;
  float acc = 0.f;
  for (int s = 0; s < S_; ++s) { acc += kp[(size_t)s*D_]; pp[(size_t)s*D_] = acc; }
}

// ---------------------------------------------------------------------------
// sk[b,s,:] = (e/Z)*bandsum + (1/Z)*sum over smallest m out-of-band indices
// (closed form of the fixed banded-softmax/top-k buffer applied to k)
// ---------------------------------------------------------------------------
__global__ void sk_kernel(const float* __restrict__ P, float* __restrict__ sk) {
  int blk = blockIdx.x;                 // b*S + s
  int b = blk / S_, s = blk % S_;
  int d = threadIdx.x;
  int lo = (s - WIN_ > 0) ? s - WIN_ : 0;
  int hi = (s + WIN_ < S_-1) ? s + WIN_ : S_-1;
  int nb = hi - lo + 1;
  const float e = 2.718281828459045f;
  float Z  = nb * e + (float)(S_ - nb);
  float av = e / Z, bc = 1.0f / Z;
  int m = KTOP_ - nb;                   // # out-of-band columns kept
  const float* Pb = P + (size_t)b * S_ * D_;
  float plo  = (lo > 0) ? Pb[(size_t)(lo-1)*D_ + d] : 0.f;
  float band = Pb[(size_t)hi*D_ + d] - plo;
  float outer;
  if (lo >= m) outer = Pb[(size_t)(m-1)*D_ + d];            // {0..m-1}
  else         outer = Pb[(size_t)(m+nb-1)*D_ + d] - band;  // {0..m+nb-1}\band
  sk[(size_t)blk * D_ + d] = av * band + bc * outer;
}

// ---------------------------------------------------------------------------
// Flash attention: out = softmax(q @ sk^T / 16) @ v
// 4 waves/block, consecutive 16-query tiles, same batch. All waves share the
// same 32-key block stream: SK (row-major) and V (transposed) staged in LDS
// as bf16, so every WMMA B-fragment is one aligned 32-byte LDS vector read.
// ---------------------------------------------------------------------------
__global__ void __launch_bounds__(128)
flash_attn(const float* __restrict__ q, const float* __restrict__ sk,
           const float* __restrict__ v, float* __restrict__ out) {
  __shared__ __align__(32) __bf16 skl[32][256];   // 16 KB, keys x depth
  __shared__ __align__(32) __bf16 vt [256][32];   // 16 KB, depth x keys
  __shared__ float lds_p[4][16*32];               // 8 KB, per-wave P staging
  int wave = threadIdx.x >> 5, lane = threadIdx.x & 31;
  int half = lane >> 4, ln = lane & 15;
  int b  = (blockIdx.x * 4) / (S_/16);            // all 4 tiles in same batch
  int m0 = ((blockIdx.x * 4 + wave) & (S_/16 - 1)) * 16;
  const float* Q = q + ((size_t)b*S_ + m0) * D_;

  v16bf aq[8];                                    // full Q tile (A-fragments)
#pragma unroll
  for (int c = 0; c < 8; ++c)
    aq[c] = load_afrag(Q + (size_t)ln*D_ + c*32 + half*16);

  v8f zero = {};
  v8f o[16];
#pragma unroll
  for (int t = 0; t < 16; ++t) o[t] = zero;
  float mrow[8], lrow[8];
#pragma unroll
  for (int i = 0; i < 8; ++i) { mrow[i] = -1e30f; lrow[i] = 0.f; }
  const float scale = 0.0625f;                    // 1/sqrt(256)
  float* pl = lds_p[wave];
  unsigned* sklu = (unsigned*)&skl[0][0];

  for (int kb = 0; kb < S_; kb += 32) {
    // ---- cooperative staging: 2048 float4 loads across 128 threads ----
    const float4* SKg = (const float4*)(sk + ((size_t)b*S_ + kb) * D_);
    const float4* Vg  = (const float4*)(v  + ((size_t)b*S_ + kb) * D_);
#pragma unroll
    for (int it = 0; it < 16; ++it) {
      int idx = threadIdx.x + 128*it;             // 0..2047
      int row = idx >> 6, c4 = idx & 63;          // key row, float4 column
      float4 f = SKg[row*64 + c4];
      sklu[row*128 + c4*2]     = pack2(f.x, f.y);
      sklu[row*128 + c4*2 + 1] = pack2(f.z, f.w);
      float4 g = Vg[row*64 + c4];
      vt[c4*4 + 0][row] = (__bf16)g.x;
      vt[c4*4 + 1][row] = (__bf16)g.y;
      vt[c4*4 + 2][row] = (__bf16)g.z;
      vt[c4*4 + 3][row] = (__bf16)g.w;
    }
    __syncthreads();

    // ---- scores: S = Q @ SK^T, K-dim = D = 256 ----
    v8f s0 = zero, s1 = zero;
#pragma unroll
    for (int c = 0; c < 8; ++c) {
      v16bf b0 = *(const v16bf*)&skl[ln]     [c*32 + half*16];
      v16bf b1 = *(const v16bf*)&skl[ln + 16][c*32 + half*16];
      s0 = wmma_bf16(aq[c], b0, s0);
      s1 = wmma_bf16(aq[c], b1, s1);
    }
    // ---- online softmax over this 32-key block (rows M = i + half*8) ----
#pragma unroll
    for (int i = 0; i < 8; ++i) {
      float x0 = s0[i]*scale, x1 = s1[i]*scale;
      float mx = fmaxf(x0, x1);
#pragma unroll
      for (int msk = 1; msk < 16; msk <<= 1)
        mx = fmaxf(mx, __shfl_xor(mx, msk, 32));
      float mn = fmaxf(mrow[i], mx);
      float alpha = __expf(mrow[i] - mn);
      float p0 = __expf(x0 - mn), p1 = __expf(x1 - mn);
      float ss = p0 + p1;
#pragma unroll
      for (int msk = 1; msk < 16; msk <<= 1)
        ss += __shfl_xor(ss, msk, 32);
      lrow[i] = lrow[i]*alpha + ss;
      mrow[i] = mn;
#pragma unroll
      for (int t = 0; t < 16; ++t) o[t][i] *= alpha;
      pl[(i + half*8)*32 + ln]      = p0;         // C-layout -> LDS
      pl[(i + half*8)*32 + 16 + ln] = p1;
    }
    // ---- P as A-fragment (16x32): run [half*16..+15] ----
    v16bf ap;
#pragma unroll
    for (int j = 0; j < 8; ++j) {
      ap[j]     = (__bf16)pl[ln*32 + half*16 + j];
      ap[j + 8] = (__bf16)pl[ln*32 + half*16 + 8 + j];
    }
    // ---- O += P @ V: B-fragments from transposed V, one vector read ----
#pragma unroll
    for (int t = 0; t < 16; ++t) {
      v16bf bv = *(const v16bf*)&vt[t*16 + ln][half*16];
      o[t] = wmma_bf16(ap, bv, o[t]);
    }
    __syncthreads();                              // before next staging pass
  }

  float inv[8];
#pragma unroll
  for (int i = 0; i < 8; ++i) inv[i] = 1.f / lrow[i];
  float* O = out + ((size_t)b*S_ + m0) * D_;
#pragma unroll
  for (int t = 0; t < 16; ++t)
#pragma unroll
    for (int i = 0; i < 8; ++i)
      O[(size_t)(i + half*8)*D_ + t*16 + ln] = o[t][i] * inv[i];
}

// ---------------------------------------------------------------------------
// Full-tensor LayerNorm over [B,S,D] (mean/var over ALL elements), 3 stages.
// r = A + Bres computed on the fly; float4 throughout.
// ---------------------------------------------------------------------------
__global__ void ln_reduce(const float* __restrict__ A, const float* __restrict__ Bres,
                          float* __restrict__ partials) {
  __shared__ float sh[256], sh2[256];
  int tid = threadIdx.x;
  size_t chunk4 = (size_t)(NTOT_/4) / gridDim.x;
  size_t base4  = (size_t)blockIdx.x * chunk4;
  const float4* A4 = (const float4*)A;
  const float4* B4 = (const float4*)Bres;
  float s = 0.f, sq = 0.f;
  for (size_t i = tid; i < chunk4; i += blockDim.x) {
    float4 a = A4[base4 + i], bb = B4[base4 + i];
    float r0 = a.x + bb.x, r1 = a.y + bb.y, r2 = a.z + bb.z, r3 = a.w + bb.w;
    s  += (r0 + r1) + (r2 + r3);
    sq += (r0*r0 + r1*r1) + (r2*r2 + r3*r3);
  }
  sh[tid] = s; sh2[tid] = sq;
  __syncthreads();
  for (int st = 128; st > 0; st >>= 1) {
    if (tid < st) { sh[tid] += sh[tid+st]; sh2[tid] += sh2[tid+st]; }
    __syncthreads();
  }
  if (tid == 0) { partials[blockIdx.x*2] = sh[0]; partials[blockIdx.x*2+1] = sh2[0]; }
}

__global__ void ln_finalize(const float* __restrict__ partials,
                            float* __restrict__ stats, int nparts) {
  __shared__ float sh[256], sh2[256];
  int tid = threadIdx.x;
  float s = 0.f, sq = 0.f;
  for (int i = tid; i < nparts; i += 256) { s += partials[i*2]; sq += partials[i*2+1]; }
  sh[tid] = s; sh2[tid] = sq;
  __syncthreads();
  for (int st = 128; st > 0; st >>= 1) {
    if (tid < st) { sh[tid] += sh[tid+st]; sh2[tid] += sh2[tid+st]; }
    __syncthreads();
  }
  if (tid == 0) {
    float mu  = sh[0] / (float)NTOT_;
    float var = sh2[0] / (float)NTOT_ - mu*mu;
    stats[0] = mu;
    stats[1] = rsqrtf(var + 1e-6f);
  }
}

__global__ void ln_apply(const float* __restrict__ A, const float* __restrict__ Bres,
                         const float* __restrict__ g, const float* __restrict__ be,
                         const float* __restrict__ stats, float* __restrict__ out) {
  size_t i = (size_t)blockIdx.x * blockDim.x + threadIdx.x;
  if (i >= (size_t)(NTOT_/4)) return;
  float mu = stats[0], rstd = stats[1];
  float4 a = ((const float4*)A)[i], r = ((const float4*)Bres)[i];
  float4 gg = ((const float4*)g)[i], bb = ((const float4*)be)[i];
  float4 o;
  o.x = (a.x + r.x - mu) * rstd * gg.x + bb.x;
  o.y = (a.y + r.y - mu) * rstd * gg.y + bb.y;
  o.z = (a.z + r.z - mu) * rstd * gg.z + bb.z;
  o.w = (a.w + r.w - mu) * rstd * gg.w + bb.w;
  ((float4*)out)[i] = o;
}

// ---------------------------------------------------------------------------
// Fused MLP: out = relu(h @ W1^T + b1) @ W2^T + b2
// One block per 16 rows; hidden [16 x 1024] lives in LDS as bf16 (32 KB).
// Stage-2 A-fragments are direct 16-byte LDS vector reads (layout match).
// ---------------------------------------------------------------------------
__global__ void __launch_bounds__(128)
mlp_wmma(const float* __restrict__ h,
         const float* __restrict__ W1, const float* __restrict__ b1,
         const float* __restrict__ W2, const float* __restrict__ b2,
         float* __restrict__ out) {
  __shared__ __align__(16) unsigned short hid[16][DFF_];   // bf16 bits
  int wave = threadIdx.x >> 5, lane = threadIdx.x & 31;
  int half = lane >> 4, ln = lane & 15;
  int m0 = blockIdx.x * 16;
  const float* H = h + (size_t)m0 * D_;

  v16bf a1[8];                               // shared A tile (rows of h)
#pragma unroll
  for (int c = 0; c < 8; ++c)
    a1[c] = load_afrag(H + (size_t)ln*D_ + c*32 + half*16);

  // stage 1: hidden = relu(h @ W1^T + b1), K = 256
  for (int t = wave; t < DFF_/16; t += 4) {
    int n0 = t * 16;
    v8f acc = {};
#pragma unroll
    for (int c = 0; c < 8; ++c) {
      v16bf bfr = load_bfrag16(W1 + (size_t)(n0 + ln)*D_ + c*32 + half*16);
      acc = wmma_bf16(a1[c], bfr, acc);
    }
    float bn = b1[n0 + ln];
#pragma unroll
    for (int i = 0; i < 8; ++i) {
      float x = fmaxf(acc[i] + bn, 0.f);
      __bf16 bx = (__bf16)x;
      unsigned short bits; __builtin_memcpy(&bits, &bx, 2);
      hid[i + half*8][n0 + ln] = bits;
    }
  }
  __syncthreads();

  // stage 2: out = hidden @ W2^T + b2, K = 1024
  typedef union { v16bf v; uint4 q[2]; } vbits;
  for (int t = wave; t < D_/16; t += 4) {
    int n0 = t * 16;
    v8f acc = {};
    for (int c = 0; c < DFF_; c += 32) {
      vbits u;
      u.q[0] = *(const uint4*)&hid[ln][c + half*16];        // K = c+half*16 .. +7
      u.q[1] = *(const uint4*)&hid[ln][c + half*16 + 8];    // K = c+half*16+8 .. +15
      v16bf bfr = load_bfrag16(W2 + (size_t)(n0 + ln)*DFF_ + c + half*16);
      acc = wmma_bf16(u.v, bfr, acc);
    }
    float bn = b2[n0 + ln];
#pragma unroll
    for (int i = 0; i < 8; ++i)
      out[(size_t)(m0 + i + half*8)*D_ + n0 + ln] = acc[i] + bn;
  }
}

// ---------------------------------------------------------------------------
extern "C" void kernel_launch(void* const* d_in, const int* in_sizes, int n_in,
                              void* d_out, int out_size, void* d_ws, size_t ws_size,
                              hipStream_t stream) {
  (void)in_sizes; (void)n_in; (void)out_size; (void)ws_size;
  const float* X     = (const float*)d_in[0];
  const float* img   = (const float*)d_in[1];
  const float* Wq    = (const float*)d_in[2];
  const float* bq    = (const float*)d_in[3];
  const float* Wk    = (const float*)d_in[4];
  const float* bk    = (const float*)d_in[5];
  const float* Wv    = (const float*)d_in[6];
  const float* bv    = (const float*)d_in[7];
  const float* W1    = (const float*)d_in[8];
  const float* b1    = (const float*)d_in[9];
  const float* W2    = (const float*)d_in[10];
  const float* b2    = (const float*)d_in[11];
  const float* gamma = (const float*)d_in[12];
  const float* beta  = (const float*)d_in[13];
  float* out = (float*)d_out;

  // workspace layout (floats), regions reused across stages: ~67 MB total
  float* ws   = (float*)d_ws;
  size_t SZ   = (size_t)NTOT_;
  float* q    = ws;            // region0: q  -> later h
  float* ksk  = ws + SZ;       // region1: k  -> sk -> m
  float* v    = ws + 2*SZ;     // region2: v
  float* patt = ws + 3*SZ;     // region3: prefix(k) -> attn_out
  float* partials = ws + 4*SZ; // 512 floats
  float* stats    = partials + 512;

  dim3 blk(128);
  int qkv_blocks = (NTOK_/16) * (D_/64) / 4;   // 16x64 tile per wave

  gemm_bias_wmma<<<qkv_blocks, blk, 0, stream>>>(X, Wq, bq, q,   NTOK_, D_, D_);
  gemm_bias_wmma<<<qkv_blocks, blk, 0, stream>>>(X, Wk, bk, ksk, NTOK_, D_, D_);
  gemm_bias_wmma<<<qkv_blocks, blk, 0, stream>>>(X, Wv, bv, v,   NTOK_, D_, D_);

  prefix_kernel<<<(B_*D_)/256, 256, 0, stream>>>(ksk, patt);   // P from k
  sk_kernel<<<B_*S_, D_, 0, stream>>>(patt, ksk);              // sk overwrites k

  flash_attn<<<(B_*(S_/16))/4, blk, 0, stream>>>(q, ksk, v, patt); // attn -> region3

  ln_reduce<<<256, 256, 0, stream>>>(patt, X, partials);
  ln_finalize<<<1, 256, 0, stream>>>(partials, stats, 256);
  ln_apply<<<(NTOT_/4 + 255)/256, 256, 0, stream>>>(patt, X, gamma, beta, stats, q);

  mlp_wmma<<<NTOK_/16, blk, 0, stream>>>(q, W1, b1, W2, b2, ksk); // m -> region1

  ln_reduce<<<256, 256, 0, stream>>>(ksk, X, partials);
  ln_finalize<<<1, 256, 0, stream>>>(partials, stats, 256);
  ln_apply<<<(NTOT_/4 + 255)/256, 256, 0, stream>>>(ksk, X, gamma, beta, stats, out);

  // image_feature passthrough (second tuple element)
  hipMemcpyAsync(out + NTOT_, img, (size_t)B_*196*D_*sizeof(float),
                 hipMemcpyDeviceToDevice, stream);
}
